// Hierarchical123GNN_37177236914924
// MI455X (gfx1250) — matrix operand order repaired and checked by reading, not compiled, not run on hardware
//
#include <hip/hip_runtime.h>

// ---------------------------------------------------------------------------
// CDNA5 (gfx1250) hierarchical 1-2-3 GNN.
// All GEMMs go through v_wmma_f32_16x16x32_f16 (f16 inputs, f32 accum).
// ---------------------------------------------------------------------------

typedef __attribute__((ext_vector_type(16))) _Float16 v16h;
typedef __attribute__((ext_vector_type(8)))  _Float16 v8h;
typedef __attribute__((ext_vector_type(8)))  float    v8f;

#define HID 64

// ---------------------------------------------------------------------------
// Dual-output WMMA GEMM:  nb = A @ W1^T,  msg = A @ W2^T
// A:  M x Kpad  (f16, row-major, Kpad % 32 == 0)
// W1/W2: 64 x Kpad (f16, row-major; rows are output features = WMMA "B" N dim)
// One wave computes a 16x64 tile for both outputs; 128-thread block = 64 rows.
// ---------------------------------------------------------------------------
__global__ __launch_bounds__(128) void gemm_dual_wmma(
    const _Float16* __restrict__ A, int Kpad,
    const _Float16* __restrict__ W1f, const _Float16* __restrict__ W2f,
    float* __restrict__ nb, float* __restrict__ msg, int M)
{
    const int wave   = threadIdx.x >> 5;
    const int lane   = threadIdx.x & 31;
    const int row0   = (blockIdx.x * 4 + wave) * 16;
    if (row0 >= M) return;                 // wave-uniform guard
    const int laneLo = lane & 15;
    const int laneHi = lane >> 4;

    const _Float16* arow = A + (size_t)(row0 + laneLo) * Kpad;

    v8f acc1[4] = {};                      // A@W1^T, 4 column tiles of 16
    v8f acc2[4] = {};                      // A@W2^T

    for (int k0 = 0; k0 < Kpad; k0 += 32) {
        // A tile (16x32): lane m holds row m, K 0..15 contiguous; lane m+16 -> K 16..31
        v8h alo = *(const v8h*)(arow + k0 + laneHi * 16);
        v8h ahi = *(const v8h*)(arow + k0 + laneHi * 16 + 8);
        v16h a;
        #pragma unroll
        for (int i = 0; i < 8; ++i) { a[i] = alo[i]; a[i + 8] = ahi[i]; }

        #pragma unroll
        for (int ct = 0; ct < 4; ++ct) {
            // B tile (32x16): lane n holds col n, K 0..15 contiguous; lane n+16 -> K 16..31
            const _Float16* b1p = W1f + (size_t)(ct * 16 + laneLo) * Kpad + k0 + laneHi * 16;
            v8h b1lo = *(const v8h*)(b1p);
            v8h b1hi = *(const v8h*)(b1p + 8);
            v16h b1;
            #pragma unroll
            for (int i = 0; i < 8; ++i) { b1[i] = b1lo[i]; b1[i + 8] = b1hi[i]; }
            acc1[ct] = __builtin_amdgcn_wmma_f32_16x16x32_f16(
                false, a, false, b1, (short)0, acc1[ct], false, false);

            const _Float16* b2p = W2f + (size_t)(ct * 16 + laneLo) * Kpad + k0 + laneHi * 16;
            v8h b2lo = *(const v8h*)(b2p);
            v8h b2hi = *(const v8h*)(b2p + 8);
            v16h b2;
            #pragma unroll
            for (int i = 0; i < 8; ++i) { b2[i] = b2lo[i]; b2[i + 8] = b2hi[i]; }
            acc2[ct] = __builtin_amdgcn_wmma_f32_16x16x32_f16(
                false, a, false, b2, (short)0, acc2[ct], false, false);
        }
    }

    // C layout: VGPR i, lanes 0-15 -> row i, lanes 16-31 -> row i+8; col = lane&15
    #pragma unroll
    for (int ct = 0; ct < 4; ++ct) {
        #pragma unroll
        for (int i = 0; i < 8; ++i) {
            size_t idx = (size_t)(row0 + laneHi * 8 + i) * HID + ct * 16 + laneLo;
            nb[idx]  = acc1[ct][i];
            msg[idx] = acc2[ct][i];
        }
    }
}

// ---------------------------------------------------------------------------
// Edge scatter: nb[dst[e]] += msg[src[e]]   (8 features per thread)
// ---------------------------------------------------------------------------
__global__ void scatter_add_kernel(const int* __restrict__ edges, int E,
                                   const float* __restrict__ msg,
                                   float* __restrict__ nb)
{
    long long t = (long long)blockIdx.x * blockDim.x + threadIdx.x;
    long long total = (long long)E * 8;
    if (t >= total) return;
    int e  = (int)(t >> 3);
    int fg = (int)(t & 7) * 8;
    int s = edges[e];
    int d = edges[E + e];
    const float* mp = msg + (size_t)s * HID + fg;
    float*       np = nb  + (size_t)d * HID + fg;
    #pragma unroll
    for (int i = 0; i < 8; ++i)
        __hip_atomic_fetch_add(np + i, mp[i], __ATOMIC_RELAXED, __HIP_MEMORY_SCOPE_AGENT);
}

// ---------------------------------------------------------------------------
// h = relu(nb); write f32 copy (for gathers / segment sums) + f16 copy (next GEMM A)
// ---------------------------------------------------------------------------
__global__ void relu_store_kernel(const float* __restrict__ nb,
                                  float* __restrict__ hf32,
                                  _Float16* __restrict__ hf16,
                                  long long count)
{
    long long t = (long long)blockIdx.x * blockDim.x + threadIdx.x;
    if (t >= count) return;
    float v = nb[t];
    v = v > 0.f ? v : 0.f;
    hf32[t] = v;
    hf16[t] = (_Float16)v;
}

// f32 -> f16 straight conversion (for input features x)
__global__ void conv_f16_kernel(const float* __restrict__ in,
                                _Float16* __restrict__ out, long long count)
{
    long long t = (long long)blockIdx.x * blockDim.x + threadIdx.x;
    if (t >= count) return;
    out[t] = (_Float16)in[t];
}

// Weight f32 (Nout x Kin) -> f16 (Nout x Kpad), zero-padded K tail
__global__ void conv_w_kernel(const float* __restrict__ W,
                              _Float16* __restrict__ out,
                              int Kin, int Kpad, int total)
{
    int t = blockIdx.x * blockDim.x + threadIdx.x;
    if (t >= total) return;
    int o = t / Kpad, k = t - o * Kpad;
    out[t] = (_Float16)(k < Kin ? W[(size_t)o * Kin + k] : 0.f);
}

// h2_in[r] = concat(h1[gu2[r]], h1[gv2[r]], iso2[r], zeros) -> Kpad=160 f16
__global__ void build_h2_kernel(const float* __restrict__ h1,
                                const int* __restrict__ gu2,
                                const int* __restrict__ gv2,
                                const float* __restrict__ iso2,
                                _Float16* __restrict__ out, int n2)
{
    long long t = (long long)blockIdx.x * blockDim.x + threadIdx.x;
    long long total = (long long)n2 * 160;
    if (t >= total) return;
    int r = (int)(t / 160), k = (int)(t - (long long)r * 160);
    float v = 0.f;
    if (k < 64)        v = h1[(size_t)gu2[r] * HID + k];
    else if (k < 128)  v = h1[(size_t)gv2[r] * HID + (k - 64)];
    else if (k == 128) v = iso2[r];
    out[t] = (_Float16)v;
}

// h3_in[r] = concat(h1[ga3[r]], h1[gb3[r]], h1[gc3[r]], iso3[r,0..3], zeros) -> Kpad=224 f16
__global__ void build_h3_kernel(const float* __restrict__ h1,
                                const int* __restrict__ ga3,
                                const int* __restrict__ gb3,
                                const int* __restrict__ gc3,
                                const float* __restrict__ iso3,
                                _Float16* __restrict__ out, int n3)
{
    long long t = (long long)blockIdx.x * blockDim.x + threadIdx.x;
    long long total = (long long)n3 * 224;
    if (t >= total) return;
    int r = (int)(t / 224), k = (int)(t - (long long)r * 224);
    float v = 0.f;
    if (k < 64)        v = h1[(size_t)ga3[r] * HID + k];
    else if (k < 128)  v = h1[(size_t)gb3[r] * HID + (k - 64)];
    else if (k < 192)  v = h1[(size_t)gc3[r] * HID + (k - 128)];
    else if (k < 196)  v = iso3[(size_t)r * 4 + (k - 192)];
    out[t] = (_Float16)v;
}

// Per-graph sum of contiguous row blocks -> comb[g, combOff:combOff+64]
__global__ void seg_sum_kernel(const float* __restrict__ h,
                               float* __restrict__ comb,
                               int rowsPerG, int combOff)
{
    int g = blockIdx.x;
    int f = threadIdx.x;                   // 64 threads = 64 features
    const float* base = h + (size_t)g * rowsPerG * HID + f;
    float s = 0.f;
    for (int r = 0; r < rowsPerG; ++r) s += base[(size_t)r * HID];
    comb[(size_t)g * 192 + combOff + f] = s;
}

// Classifier: hid = relu(comb @ cW1^T + cb1); out = hid @ cW2^T + cb2
__global__ void classifier_kernel(const float* __restrict__ comb,
                                  const float* __restrict__ cW1,
                                  const float* __restrict__ cb1,
                                  const float* __restrict__ cW2,
                                  const float* __restrict__ cb2,
                                  float* __restrict__ out)
{
    __shared__ float hid[HID];
    int g = blockIdx.x;
    int t = threadIdx.x;                   // 64 threads
    const float* c = comb + (size_t)g * 192;
    float s = cb1[t];
    for (int k = 0; k < 192; ++k) s += c[k] * cW1[(size_t)t * 192 + k];
    hid[t] = s > 0.f ? s : 0.f;
    __syncthreads();
    if (t < 10) {
        float o = cb2[t];
        for (int k = 0; k < HID; ++k) o += hid[k] * cW2[(size_t)t * HID + k];
        out[(size_t)g * 10 + t] = o;
    }
}

// ---------------------------------------------------------------------------
// Host orchestration
// ---------------------------------------------------------------------------
extern "C" void kernel_launch(void* const* d_in, const int* in_sizes, int n_in,
                              void* d_out, int out_size, void* d_ws, size_t ws_size,
                              hipStream_t stream)
{
    (void)n_in; (void)ws_size;

    // ---- inputs (setup_inputs dict order) ----
    const float* x      = (const float*)d_in[0];
    const int*   ei     = (const int*)d_in[1];
    const int*   gu2    = (const int*)d_in[3];
    const int*   gv2    = (const int*)d_in[4];
    const float* iso2   = (const float*)d_in[5];
    const int*   e2     = (const int*)d_in[6];
    const int*   ga3    = (const int*)d_in[8];
    const int*   gb3    = (const int*)d_in[9];
    const int*   gc3    = (const int*)d_in[10];
    const float* iso3   = (const float*)d_in[11];
    const int*   e3     = (const int*)d_in[12];
    const float* cW1    = (const float*)d_in[28];
    const float* cb1    = (const float*)d_in[29];
    const float* cW2    = (const float*)d_in[30];
    const float* cb2    = (const float*)d_in[31];

    const int n1 = in_sizes[2];        // 2560 nodes
    const int E1 = in_sizes[1] / 2;
    const int n2 = in_sizes[3];        // 24320 2-tuples
    const int E2 = in_sizes[6] / 2;
    const int n3 = in_sizes[8];        // 145920 3-tuples
    const int E3 = in_sizes[12] / 2;
    const int Gg = out_size / 10;      // 128 graphs

    // ---- workspace bump allocator (256B aligned) ----
    char* wsp = (char*)d_ws;
    auto alloc = [&](size_t bytes) -> void* {
        void* r = (void*)wsp;
        wsp += (bytes + 255) & ~(size_t)255;
        return r;
    };

    _Float16* A16 = (_Float16*)alloc(sizeof(_Float16) * (size_t)n3 * 224);  // max A
    float* nb   = (float*)alloc(sizeof(float) * (size_t)n3 * HID);
    float* msg  = (float*)alloc(sizeof(float) * (size_t)n3 * HID);
    float* h1   = (float*)alloc(sizeof(float) * (size_t)n1 * HID);
    float* h2   = (float*)alloc(sizeof(float) * (size_t)n2 * HID);
    float* h3   = (float*)alloc(sizeof(float) * (size_t)n3 * HID);
    float* comb = (float*)alloc(sizeof(float) * (size_t)Gg * 192);

    // f16 weight buffers, zero-padded K
    auto allocw = [&](int kpad) { return (_Float16*)alloc(sizeof(_Float16) * 64 * (size_t)kpad); };
    _Float16* w[14];
    const int  widx [14] = {14, 15, 16, 17, 18, 19, 20, 21, 22, 23, 24, 25, 26, 27};
    const int  wkin [14] = {32, 32, 64, 64, 64, 64, 129, 129, 64, 64, 196, 196, 64, 64};
    const int  wkpad[14] = {32, 32, 64, 64, 64, 64, 160, 160, 64, 64, 224, 224, 64, 64};
    for (int i = 0; i < 14; ++i) w[i] = allocw(wkpad[i]);

    auto cdiv = [](long long a, long long b) { return (int)((a + b - 1) / b); };

    // ---- convert weights to padded f16 ----
    for (int i = 0; i < 14; ++i) {
        int tot = 64 * wkpad[i];
        conv_w_kernel<<<cdiv(tot, 256), 256, 0, stream>>>(
            (const float*)d_in[widx[i]], w[i], wkin[i], wkpad[i], tot);
    }

    auto run_layer = [&](int M, int Kpad, _Float16* W1f, _Float16* W2f,
                         const int* edges, int E, float* hf32) {
        gemm_dual_wmma<<<cdiv(M, 64), 128, 0, stream>>>(A16, Kpad, W1f, W2f, nb, msg, M);
        scatter_add_kernel<<<cdiv((long long)E * 8, 256), 256, 0, stream>>>(edges, E, msg, nb);
        relu_store_kernel<<<cdiv((long long)M * HID, 256), 256, 0, stream>>>(
            nb, hf32, A16, (long long)M * HID);
    };

    // ---- stage 1: node GNN (3 layers) ----
    conv_f16_kernel<<<cdiv((long long)n1 * 32, 256), 256, 0, stream>>>(x, A16, (long long)n1 * 32);
    run_layer(n1, 32, w[0], w[1], ei, E1, h1);
    run_layer(n1, 64, w[2], w[3], ei, E1, h1);
    run_layer(n1, 64, w[4], w[5], ei, E1, h1);
    seg_sum_kernel<<<Gg, 64, 0, stream>>>(h1, comb, n1 / Gg, 0);

    // ---- stage 2: 2-tuple GNN (2 layers) ----
    build_h2_kernel<<<cdiv((long long)n2 * 160, 256), 256, 0, stream>>>(h1, gu2, gv2, iso2, A16, n2);
    run_layer(n2, 160, w[6], w[7], e2, E2, h2);
    run_layer(n2, 64,  w[8], w[9], e2, E2, h2);
    seg_sum_kernel<<<Gg, 64, 0, stream>>>(h2, comb, n2 / Gg, 64);

    // ---- stage 3: 3-tuple GNN (2 layers) ----
    build_h3_kernel<<<cdiv((long long)n3 * 224, 256), 256, 0, stream>>>(h1, ga3, gb3, gc3, iso3, A16, n3);
    run_layer(n3, 224, w[10], w[11], e3, E3, h3);
    run_layer(n3, 64,  w[12], w[13], e3, E3, h3);
    seg_sum_kernel<<<Gg, 64, 0, stream>>>(h3, comb, n3 / Gg, 128);

    // ---- classifier ----
    classifier_kernel<<<Gg, 64, 0, stream>>>(comb, cW1, cb1, cW2, cb2, (float*)d_out);
}
